// LambdaLayer_78408922956117
// MI455X (gfx1250) — matrix operand order repaired and checked by reading, not compiled, not run on hardware
//
#include <hip/hip_runtime.h>
#include <hip/hip_bf16.h>

typedef __bf16 bf16;
typedef __attribute__((ext_vector_type(16))) __bf16 v16bf;
typedef __attribute__((ext_vector_type(8)))  __bf16 v8bf;
typedef __attribute__((ext_vector_type(8)))  float  v8f;

#define BB   16
#define CC   256
#define HHW  64
#define NN   4096
#define KK_  16
#define HH_  4
#define UU   4
#define RR   23
#define DVV  64
#define PP   11
#define XPAD 96   // 11 left pad + 64 data + 21 right pad
#define EPSB 1e-3f

// ---------- workspace layout (bytes) ----------
#define OFF_XT    ((size_t)0)                         // [b][n][c] bf16 : 33,554,432
#define OFF_WALL  ((size_t)33554432)                  // [384][256] bf16: 196,608
#define OFF_WPOS  ((size_t)33751040)                  // [k*4+u][23][32] bf16: 94,208
#define OFF_QF    ((size_t)33845248)                  // [b][64][4096] f32: 16,777,216
#define OFF_KF    ((size_t)50622464)                  // [b][64][4096] f32: 16,777,216
#define OFF_KSM   ((size_t)67399680)                  // [b][64][4096] bf16: 8,388,608
#define OFF_VPAD  ((size_t)75788288)                  // [b][u][dv][64][96] bf16: 50,331,648
#define OFF_LC    ((size_t)126119936)                 // [b][16][64] f32: 65,536

__device__ inline bf16 f2bf(float f) {
  unsigned u = __builtin_bit_cast(unsigned, f);
  unsigned r = u + 0x7FFFu + ((u >> 16) & 1u);
  unsigned short s = (unsigned short)(r >> 16);
  return __builtin_bit_cast(bf16, s);
}

union V16U { v16bf v; v8bf h[2]; };

// aligned (>=16B) pair loader -> A/B operand
__device__ inline v16bf ldA(const bf16* p0, const bf16* p1) {
  V16U r;
  r.h[0] = *(const v8bf*)p0;
  r.h[1] = *(const v8bf*)p1;
  return r.v;
}
// unaligned (2B) pair loader (shifted conv windows)
__device__ inline v16bf ldU(const bf16* p0, const bf16* p1) {
  V16U r;
  __builtin_memcpy(&r.h[0], p0, 16);
  __builtin_memcpy(&r.h[1], p1, 16);
  return r.v;
}

__device__ inline v8f wmma_bf16(v16bf a, v16bf b, v8f c) {
  return __builtin_amdgcn_wmma_f32_16x16x32_bf16(false, a, false, b, (short)0, c, false, false);
}

// ---------------- prep kernels ----------------
__global__ void zero_vpad_k(uint4* p) {
  size_t i = (size_t)blockIdx.x * 256 + threadIdx.x;   // 3,145,728 uint4
  p[i] = uint4{0u, 0u, 0u, 0u};
}

__global__ void pack_weights_k(const float* wq, const float* wk, const float* wv, bf16* wall) {
  int i = blockIdx.x * 256 + threadIdx.x;              // 384*256
  int row = i >> 8, c = i & 255;
  float v;
  if (row < 64)       v = wq[row * 256 + c];
  else if (row < 128) v = wk[(row - 64) * 256 + c];
  else                v = wv[(row - 128) * 256 + c];
  wall[i] = f2bf(v);
}

__global__ void pack_wpos_k(const float* wpos, bf16* wposb) {
  int i = blockIdx.x * 256 + threadIdx.x;              // 16*4*23*32 = 47104
  int s = i & 31;
  int r = (i >> 5) % RR;
  int ku = i / (32 * RR);                               // ku = k*4+u
  float v = (s < RR) ? wpos[(ku * RR + r) * RR + s] : 0.0f;
  wposb[i] = f2bf(v);
}

__global__ void transpose_x_k(const float* x, bf16* xT) {
  __shared__ float tile[32][33];
  int bid = blockIdx.x;
  int nt = bid & 127;
  int ct = (bid >> 7) & 7;
  int b  = bid >> 10;
  int tx = threadIdx.x & 31;
  int ty = threadIdx.x >> 5;                            // 0..7
  const float* xb = x + (size_t)b * CC * NN;
  bf16* xtb = xT + (size_t)b * NN * CC;
  for (int i = 0; i < 4; i++) {
    int c = ct * 32 + ty + i * 8;
    tile[ty + i * 8][tx] = xb[(size_t)c * NN + nt * 32 + tx];
  }
  __syncthreads();
  for (int i = 0; i < 4; i++) {
    int n = nt * 32 + ty + i * 8;
    xtb[(size_t)n * CC + ct * 32 + tx] = f2bf(tile[tx][ty + i * 8]);
  }
}

// ---------------- projection GEMM (WMMA bf16) ----------------
__global__ void proj_gemm_k(const bf16* xT, const bf16* wall,
                            const float* gq, const float* bq, const float* mq, const float* vq,
                            const float* gv, const float* bv, const float* mv, const float* vv,
                            float* qf, float* kf, bf16* vpad) {
  int w = blockIdx.x * 8 + (threadIdx.x >> 5);          // 98304 waves
  int lane = threadIdx.x & 31;
  int nt = w & 255;
  int ot = (w >> 8) % 24;
  int b  = w / 6144;
  int col = lane & 15, hi = lane >> 4;
  int n = nt * 16 + col;
  const bf16* arow = wall + (size_t)(ot * 16 + col) * 256;
  const bf16* brow = xT + ((size_t)b * NN + n) * CC;
  int kc0 = hi ? 16 : 0;
  int kb0 = hi ? 16 : 0;
  v8f acc = {};
  for (int ks = 0; ks < 256; ks += 32) {
    v16bf a = ldA(arow + ks + kc0, arow + ks + kc0 + 8);
    v16bf bm = ldA(brow + ks + kb0, brow + ks + kb0 + 8);
    acc = wmma_bf16(a, bm, acc);
  }
  int y = n >> 6, xx = n & 63;
  for (int j = 0; j < 8; j++) {
    int oc = ot * 16 + j + hi * 8;
    float val = acc[j];
    if (oc < 64) {
      float inv = gq[oc] * rsqrtf(vq[oc] + EPSB);
      qf[((size_t)b * 64 + oc) * NN + n] = val * inv + (bq[oc] - mq[oc] * inv);
    } else if (oc < 128) {
      kf[((size_t)b * 64 + (oc - 64)) * NN + n] = val;
    } else {
      int ocv = oc - 128;
      float inv = gv[ocv] * rsqrtf(vv[ocv] + EPSB);
      float r = val * inv + (bv[ocv] - mv[ocv] * inv);
      int u = ocv >> 6, dv = ocv & 63;
      vpad[((((size_t)b * 4 + u) * 64 + dv) * 64 + y) * XPAD + PP + xx] = f2bf(r);
    }
  }
}

// ---------------- softmax over m ----------------
__global__ void softmax_k(const float* kf, bf16* ksm) {
  __shared__ float red[256];
  int row = blockIdx.x;                                  // 1024 rows
  const float* src = kf + (size_t)row * NN;
  bf16* dst = ksm + (size_t)row * NN;
  int t = threadIdx.x;
  float m = -1e30f;
  for (int i = t; i < NN; i += 256) m = fmaxf(m, src[i]);
  red[t] = m; __syncthreads();
  for (int s = 128; s > 0; s >>= 1) { if (t < s) red[t] = fmaxf(red[t], red[t + s]); __syncthreads(); }
  m = red[0]; __syncthreads();
  float sum = 0.f;
  for (int i = t; i < NN; i += 256) sum += __expf(src[i] - m);
  red[t] = sum; __syncthreads();
  for (int s = 128; s > 0; s >>= 1) { if (t < s) red[t] += red[t + s]; __syncthreads(); }
  float inv = 1.0f / red[0];
  for (int i = t; i < NN; i += 256) dst[i] = f2bf(__expf(src[i] - m) * inv);
}

// ---------------- Lc = softmax(k) . v^T (WMMA bf16) ----------------
__global__ void lc_gemm_k(const bf16* ksm, const bf16* vpad, float* lc) {
  int w = blockIdx.x;                                    // 64 waves, block=32
  int dvt = w & 3;
  int b = w >> 2;
  int lane = threadIdx.x & 31;
  int col = lane & 15, hi = lane >> 4;
  int kc0 = hi ? 16 : 0, kb0 = hi ? 16 : 0;
  v8f acc = {};
  for (int u = 0; u < 4; u++) {
    const bf16* arow = ksm + ((size_t)b * 64 + u * 16 + col) * NN;
    const bf16* vb = vpad + (((size_t)b * 4 + u) * 64 + dvt * 16 + col) * 64 * XPAD;
    for (int y = 0; y < 64; y++) {
      const bf16* vr = vb + y * XPAD + PP;               // row start (2B aligned only)
      for (int c2 = 0; c2 < 2; c2++) {
        int ms = y * 64 + c2 * 32;
        v16bf a = ldA(arow + ms + kc0, arow + ms + kc0 + 8);
        v16bf bm = ldU(vr + c2 * 32 + kb0, vr + c2 * 32 + kb0 + 8);
        acc = wmma_bf16(a, bm, acc);
      }
    }
  }
  for (int j = 0; j < 8; j++) {
    int kk = j + hi * 8;
    lc[((size_t)b * 16 + kk) * 64 + dvt * 16 + col] = acc[j];
  }
}

// ---------------- fused position-conv + Yc + Yp ----------------
__global__ void lambda_out_k(const bf16* vpad, const bf16* wposb, const float* qf,
                             const float* lc, const float* b_pos, float* out) {
  int w = blockIdx.x * 8 + (threadIdx.x >> 5);           // 65536 waves
  int lane = threadIdx.x & 31;
  int dvg = w & 15;
  int xt = (w >> 4) & 3;
  int y  = (w >> 6) & 63;
  int b  = w >> 12;
  int col = lane & 15, hi = lane >> 4;
  int xbase = xt * 16;
  int ka0 = hi ? 16 : 0, kb0 = hi ? 16 : 0;
  v8f acc[4] = {{}, {}, {}, {}};

  const size_t dvStride = (size_t)64 * XPAD;
  for (int r = 0; r < RR; r++) {
    int yy = y + r - PP;
    if (yy < 0 || yy >= 64) continue;
    for (int u = 0; u < 4; u++) {
      const bf16* ar = wposb + (((size_t)col * 4 + u) * RR + r) * 32;
      v16bf a = ldA(ar + ka0, ar + ka0 + 8);
      const bf16* vb = vpad + ((((size_t)b * 4 + u) * 64 + dvg * 4) * 64 + yy) * XPAD;
      const bf16* p = vb + xbase + col + kb0;            // shifted window: 2B aligned
      __builtin_prefetch(p + dvStride * 4, 0, 1);
#pragma unroll
      for (int d = 0; d < 4; d++) {
        v16bf bm = ldU(p, p + 8);
        acc[d] = wmma_bf16(a, bm, acc[d]);
        p += dvStride;
      }
    }
  }

  // contract k-dim with q; fold Yc via Lc + b_pos added to accumulator
  int n = y * 64 + xbase + col;
  float qv[4][8];
#pragma unroll
  for (int h = 0; h < 4; h++)
#pragma unroll
    for (int j = 0; j < 8; j++)
      qv[h][j] = qf[(((size_t)b * 4 + h) * 16 + j + hi * 8) * NN + n];

#pragma unroll
  for (int d = 0; d < 4; d++) {
    int dv = dvg * 4 + d;
    float lca[8];
#pragma unroll
    for (int j = 0; j < 8; j++) {
      int kkj = j + hi * 8;
      lca[j] = lc[((size_t)b * 16 + kkj) * 64 + dv] + b_pos[kkj];
    }
#pragma unroll
    for (int h = 0; h < 4; h++) {
      float pr = 0.f;
#pragma unroll
      for (int j = 0; j < 8; j++) pr += qv[h][j] * (acc[d][j] + lca[j]);
      float tot = pr + __shfl_xor(pr, 16);
      if (hi == 0)
        out[(((size_t)b * 4 + h) * 64 + dv) * NN + n] = tot;
    }
  }
}

extern "C" void kernel_launch(void* const* d_in, const int* in_sizes, int n_in,
                              void* d_out, int out_size, void* d_ws, size_t ws_size,
                              hipStream_t stream) {
  (void)in_sizes; (void)n_in; (void)out_size; (void)ws_size;
  const float* x     = (const float*)d_in[0];
  const float* w_q   = (const float*)d_in[1];
  const float* w_k   = (const float*)d_in[2];
  const float* w_v   = (const float*)d_in[3];
  const float* g_q   = (const float*)d_in[4];
  const float* b_q   = (const float*)d_in[5];
  const float* m_q   = (const float*)d_in[6];
  const float* v_q   = (const float*)d_in[7];
  const float* g_v   = (const float*)d_in[8];
  const float* b_v   = (const float*)d_in[9];
  const float* m_v   = (const float*)d_in[10];
  const float* v_v   = (const float*)d_in[11];
  const float* w_pos = (const float*)d_in[12];
  const float* b_pos = (const float*)d_in[13];
  float* out = (float*)d_out;

  char* ws = (char*)d_ws;
  bf16*  xT    = (bf16*)(ws + OFF_XT);
  bf16*  wall  = (bf16*)(ws + OFF_WALL);
  bf16*  wposb = (bf16*)(ws + OFF_WPOS);
  float* qf    = (float*)(ws + OFF_QF);
  float* kf    = (float*)(ws + OFF_KF);
  bf16*  ksm   = (bf16*)(ws + OFF_KSM);
  bf16*  vpad  = (bf16*)(ws + OFF_VPAD);
  float* lcbuf = (float*)(ws + OFF_LC);

  zero_vpad_k<<<12288, 256, 0, stream>>>((uint4*)vpad);
  pack_weights_k<<<384, 256, 0, stream>>>(w_q, w_k, w_v, wall);
  pack_wpos_k<<<184, 256, 0, stream>>>(w_pos, wposb);
  transpose_x_k<<<16384, 256, 0, stream>>>(x, xT);
  proj_gemm_k<<<12288, 256, 0, stream>>>(xT, wall, g_q, b_q, m_q, v_q,
                                         g_v, b_v, m_v, v_v, qf, kf, vpad);
  softmax_k<<<1024, 256, 0, stream>>>(kf, ksm);
  lc_gemm_k<<<64, 32, 0, stream>>>(ksm, vpad, lcbuf);
  lambda_out_k<<<8192, 256, 0, stream>>>(vpad, wposb, qf, lcbuf, b_pos, out);
}
